// ProteinConditionedUNet_28424093565058
// MI455X (gfx1250) — matrix operand order repaired and checked by reading, not compiled, not run on hardware
//
#include <hip/hip_runtime.h>
#include <cstdint>
#include <cstddef>

// ---------------------------------------------------------------------------
// Types for CDNA5 WMMA (gfx1250, wave32)
// ---------------------------------------------------------------------------
typedef __bf16 bf16_t;
typedef __attribute__((ext_vector_type(16))) __bf16 v16bf;
typedef __attribute__((ext_vector_type(8)))  __bf16 v8bf;
typedef __attribute__((ext_vector_type(8)))  float  v8f;
typedef __attribute__((ext_vector_type(4)))  int    int4v;

#define H_DIM 512
#define FF_DIM 2048

// Async global->LDS staging (gfx1250 ASYNCcnt path), guarded so we fall back
// to register-staged copies if the toolchain lacks the builtins.
#if defined(__AMDGCN__) && \
    __has_builtin(__builtin_amdgcn_global_load_async_to_lds_b128) && \
    __has_builtin(__builtin_amdgcn_s_wait_asynccnt)
#define USE_ASYNC_LDS 1
// builtin signature: (v4i addrspace(1)* src, v4i addrspace(3)* dst, imm offset, imm cpol)
#define GQ(p) ((__attribute__((address_space(1))) int4v*)(void*)(p))
#define LQ(p) ((__attribute__((address_space(3))) int4v*)(void*)(p))
#else
#define USE_ASYNC_LDS 0
#endif

union V16U { v16bf v; v8bf h[2]; };

__device__ __forceinline__ bf16_t f2bf(float f) {
  union { float f; uint32_t u; } v; v.f = f;
  uint32_t r = v.u + 0x7fffu + ((v.u >> 16) & 1u);   // round-to-nearest-even
  uint16_t hh = (uint16_t)(r >> 16);
  bf16_t o; __builtin_memcpy(&o, &hh, 2); return o;
}

__device__ __forceinline__ float siluf(float x) { return x / (1.0f + __expf(-x)); }
__device__ __forceinline__ float geluf(float x) {
  return 0.5f * x * (1.0f + erff(x * 0.70710678118654752440f));   // exact erf gelu
}

// ---------------------------------------------------------------------------
// WMMA GEMM: C[M x Ntot] = A[M x K](bf16, row major) * Wt[Ntot x K](bf16,
// pre-transposed so B columns are contiguous) + bias, with fused epilogue.
// Block tile 128x128, 8 waves; each wave computes 32x64 via 2x4 wmma tiles.
// K stepped by 32, LDS double-buffered; staging uses the gfx1250 async
// global->LDS engine when available (no VGPR round-trip).
// ---------------------------------------------------------------------------
#define MODE_F32        0
#define MODE_BF16       1
#define MODE_F32_RESID  2
#define MODE_BF16_GELU  3

template <int MODE>
__global__ __launch_bounds__(256) void gemm_bf16(
    const bf16_t* __restrict__ A, const bf16_t* __restrict__ Wt,
    const float* __restrict__ bias, const float* __restrict__ R,
    void* __restrict__ Cout, int Ntot, int K)
{
  __shared__ bf16_t ldsA[2][128 * 40];   // 32 K-halves per row, padded to 40
  __shared__ bf16_t ldsW[2][128 * 40];

  const int tid    = threadIdx.x;
  const int lane   = tid & 31;
  const int wave   = tid >> 5;
  const int waveM  = wave & 3;                 // 4 waves along M
  const int waveN  = wave >> 2;                // 2 waves along N
  const int blockM = blockIdx.y * 128;
  const int blockN = blockIdx.x * 128;

  // staging: 512 chunks of 16B per tile (128 rows x 4 chunks); 2 per thread
  const int c0    = tid * 2;
  const int rowS0 = c0 >> 2,        csS0 = c0 & 3;
  const int rowS1 = (c0 + 1) >> 2,  csS1 = (c0 + 1) & 3;

  v8f acc[2][4] = {};
  const int KT = K >> 5;

#if USE_ASYNC_LDS
  auto stage_async = [&](int buf, int kk) {
    __builtin_amdgcn_global_load_async_to_lds_b128(
        GQ(A  + (size_t)(blockM + rowS0) * K + kk + csS0 * 8),
        LQ(&ldsA[buf][rowS0 * 40 + csS0 * 8]), 0, 0);
    __builtin_amdgcn_global_load_async_to_lds_b128(
        GQ(A  + (size_t)(blockM + rowS1) * K + kk + csS1 * 8),
        LQ(&ldsA[buf][rowS1 * 40 + csS1 * 8]), 0, 0);
    __builtin_amdgcn_global_load_async_to_lds_b128(
        GQ(Wt + (size_t)(blockN + rowS0) * K + kk + csS0 * 8),
        LQ(&ldsW[buf][rowS0 * 40 + csS0 * 8]), 0, 0);
    __builtin_amdgcn_global_load_async_to_lds_b128(
        GQ(Wt + (size_t)(blockN + rowS1) * K + kk + csS1 * 8),
        LQ(&ldsW[buf][rowS1 * 40 + csS1 * 8]), 0, 0);
  };

  stage_async(0, 0);
  __builtin_amdgcn_s_wait_asynccnt(0);
  __syncthreads();
#else
  {
    *(v8bf*)&ldsA[0][rowS0 * 40 + csS0 * 8] = *(const v8bf*)(A  + (size_t)(blockM + rowS0) * K + csS0 * 8);
    *(v8bf*)&ldsA[0][rowS1 * 40 + csS1 * 8] = *(const v8bf*)(A  + (size_t)(blockM + rowS1) * K + csS1 * 8);
    *(v8bf*)&ldsW[0][rowS0 * 40 + csS0 * 8] = *(const v8bf*)(Wt + (size_t)(blockN + rowS0) * K + csS0 * 8);
    *(v8bf*)&ldsW[0][rowS1 * 40 + csS1 * 8] = *(const v8bf*)(Wt + (size_t)(blockN + rowS1) * K + csS1 * 8);
  }
  __syncthreads();
#endif

  const int khalf = lane >> 4;
  const int lrow  = lane & 15;

  for (int kt = 0; kt < KT; ++kt) {
    const int cur = kt & 1;
    const bool pre = (kt + 1 < KT);

#if USE_ASYNC_LDS
    if (pre) stage_async(cur ^ 1, (kt + 1) << 5);   // async copy overlaps wmma
#else
    v8bf pa0, pa1, pw0, pw1;
    if (pre) {   // prefetch next k-tile into registers (overlaps with wmma)
      const int kk = (kt + 1) << 5;
      pa0 = *(const v8bf*)(A  + (size_t)(blockM + rowS0) * K + kk + csS0 * 8);
      pa1 = *(const v8bf*)(A  + (size_t)(blockM + rowS1) * K + kk + csS1 * 8);
      pw0 = *(const v8bf*)(Wt + (size_t)(blockN + rowS0) * K + kk + csS0 * 8);
      pw1 = *(const v8bf*)(Wt + (size_t)(blockN + rowS1) * K + kk + csS1 * 8);
    }
#endif

    // A fragments: 16x32 bf16 per ISA layout (lane m=lrow, khalf selects K-halves)
    V16U af[2], bfr[4];
    #pragma unroll
    for (int mi = 0; mi < 2; ++mi) {
      const int base = (waveM * 32 + mi * 16 + lrow) * 40;
      af[mi].h[0] = *(const v8bf*)&ldsA[cur][base + khalf * 8];
      af[mi].h[1] = *(const v8bf*)&ldsA[cur][base + 16 + khalf * 8];
    }
    // B fragments: 32x16 bf16; column = lrow of Wt (row of Wt is N-index, K contiguous)
    #pragma unroll
    for (int ni = 0; ni < 4; ++ni) {
      const int base = (waveN * 64 + ni * 16 + lrow) * 40;
      bfr[ni].h[0] = *(const v8bf*)&ldsW[cur][base + khalf * 8];
      bfr[ni].h[1] = *(const v8bf*)&ldsW[cur][base + 16 + khalf * 8];
    }

    #pragma unroll
    for (int mi = 0; mi < 2; ++mi)
      #pragma unroll
      for (int ni = 0; ni < 4; ++ni)
        acc[mi][ni] = __builtin_amdgcn_wmma_f32_16x16x32_bf16(
            false, af[mi].v, false, bfr[ni].v, (short)0, acc[mi][ni], false, false);

#if USE_ASYNC_LDS
    __builtin_amdgcn_s_wait_asynccnt(0);   // our async copies into nxt done
#else
    if (pre) {
      const int nxt = cur ^ 1;
      *(v8bf*)&ldsA[nxt][rowS0 * 40 + csS0 * 8] = pa0;
      *(v8bf*)&ldsA[nxt][rowS1 * 40 + csS1 * 8] = pa1;
      *(v8bf*)&ldsW[nxt][rowS0 * 40 + csS0 * 8] = pw0;
      *(v8bf*)&ldsW[nxt][rowS1 * 40 + csS1 * 8] = pw1;
    }
#endif
    __syncthreads();
  }

  // epilogue: C/D tile layout — VGPR r holds M=r (lanes 0-15) / M=r+8 (lanes 16-31)
  const int rhi = (lane >> 4) * 8;
  #pragma unroll
  for (int mi = 0; mi < 2; ++mi) {
    #pragma unroll
    for (int ni = 0; ni < 4; ++ni) {
      const int gcol = blockN + waveN * 64 + ni * 16 + lrow;
      const float bb = bias[gcol];
      #pragma unroll
      for (int r = 0; r < 8; ++r) {
        const int grow = blockM + waveM * 32 + mi * 16 + r + rhi;
        const size_t idx = (size_t)grow * Ntot + gcol;
        float v = acc[mi][ni][r] + bb;
        if (MODE == MODE_F32)            ((float*)Cout)[idx]  = v;
        else if (MODE == MODE_BF16)      ((bf16_t*)Cout)[idx] = f2bf(v);
        else if (MODE == MODE_F32_RESID) ((float*)Cout)[idx]  = v + R[idx];
        else                             ((bf16_t*)Cout)[idx] = f2bf(geluf(v));
      }
    }
  }
}

// ---------------------------------------------------------------------------
// LayerNorm (+ optional SiLU), f32 in -> bf16 out. One wave per 512-wide row.
// ---------------------------------------------------------------------------
__global__ __launch_bounds__(256) void ln_act_kernel(
    const float* __restrict__ X, const float* __restrict__ g, const float* __restrict__ b,
    bf16_t* __restrict__ Y, int act)
{
  const int lane = threadIdx.x & 31;
  const int wave = threadIdx.x >> 5;
  const int row  = blockIdx.x * 8 + wave;
  const float* Xr = X + (size_t)row * H_DIM;
  float xv[16];
  float s = 0.f, sq = 0.f;
  #pragma unroll
  for (int i = 0; i < 16; ++i) {
    float v = Xr[lane + i * 32];
    xv[i] = v; s += v; sq += v * v;
  }
  #pragma unroll
  for (int off = 16; off > 0; off >>= 1) {
    s  += __shfl_xor(s,  off, 32);
    sq += __shfl_xor(sq, off, 32);
  }
  const float mean = s * (1.0f / H_DIM);
  const float var  = sq * (1.0f / H_DIM) - mean * mean;
  const float rstd = rsqrtf(var + 1e-5f);
  bf16_t* Yr = Y + (size_t)row * H_DIM;
  #pragma unroll
  for (int i = 0; i < 16; ++i) {
    const int c = lane + i * 32;
    float v = (xv[i] - mean) * rstd * g[c] + b[c];
    if (act) v = siluf(v);
    Yr[c] = f2bf(v);
  }
}

// time embedding first stage: e1[b,j] = silu(float(t[b]) * w1[j] + b1[j]) -> bf16
__global__ void time_e1_kernel(const int* __restrict__ t, const float* __restrict__ w1,
                               const float* __restrict__ b1, bf16_t* __restrict__ Y)
{
  const int id = blockIdx.x * blockDim.x + threadIdx.x;
  const int bi = id >> 9, j = id & 511;
  const float tf = (float)t[bi];
  Y[id] = f2bf(siluf(tf * w1[j] + b1[j]));
}

// f32 -> bf16 elementwise
__global__ void cvt_bf16_kernel(const float* __restrict__ X, bf16_t* __restrict__ Y, int n)
{
  const int id = blockIdx.x * blockDim.x + threadIdx.x;
  if (id < n) Y[id] = f2bf(X[id]);
}

// W[K x N] f32 -> Wt[N x K] bf16 (transpose + convert; weights are small)
__global__ void wtrans_kernel(const float* __restrict__ W, bf16_t* __restrict__ Wt, int K, int N)
{
  const int id = blockIdx.x * blockDim.x + threadIdx.x;
  if (id >= K * N) return;
  const int k = id / N, n = id - k * N;
  Wt[(size_t)n * K + k] = f2bf(W[id]);
}

// ---------------------------------------------------------------------------
// Host-side orchestration
// ---------------------------------------------------------------------------
extern "C" void kernel_launch(void* const* d_in, const int* in_sizes, int n_in,
                              void* d_out, int out_size, void* d_ws, size_t ws_size,
                              hipStream_t stream) {
  (void)n_in; (void)out_size; (void)ws_size;
  auto F = [&](int i) { return (const float*)d_in[i]; };

  const float* x  = F(0);
  const int*   t  = (const int*)d_in[1];
  const float* pc = F(2);
  const int Brows = in_sizes[1];          // B = 32768

  int q = 3;
  // time
  const float *tw1 = F(q++), *tb1 = F(q++), *tw2 = F(q++), *tb2 = F(q++);
  // prot
  const float *pw1 = F(q++), *pb1 = F(q++), *pg = F(q++), *pbn = F(q++), *pw2 = F(q++), *pb2 = F(q++);
  // comp
  const float *cw = F(q++), *cb = F(q++);
  // cross blocks: wv,bv,wo,bo,gf,bf,w1,b1,w2,b2,gx,bx,gc,bc
  const float *cr_wv[3], *cr_bv[3], *cr_wo[3], *cr_bo[3], *cr_gf[3], *cr_bf[3],
              *cr_w1[3], *cr_b1[3], *cr_w2[3], *cr_b2[3], *cr_gc[3], *cr_bc[3];
  for (int l = 0; l < 3; ++l) {
    cr_wv[l] = F(q++); cr_bv[l] = F(q++); cr_wo[l] = F(q++); cr_bo[l] = F(q++);
    cr_gf[l] = F(q++); cr_bf[l] = F(q++); cr_w1[l] = F(q++); cr_b1[l] = F(q++);
    cr_w2[l] = F(q++); cr_b2[l] = F(q++);
    (void)F(q++); (void)F(q++);           // gx, bx (unused: degenerate attention)
    cr_gc[l] = F(q++); cr_bc[l] = F(q++);
  }
  // self blocks: wv,bv,wo,bo,gf,bf,w1,b1,w2,b2,g,bs
  const float *sf_wv[3], *sf_bv[3], *sf_wo[3], *sf_bo[3], *sf_gf[3], *sf_bf[3],
              *sf_w1[3], *sf_b1[3], *sf_w2[3], *sf_b2[3], *sf_g[3], *sf_bs[3];
  for (int l = 0; l < 3; ++l) {
    sf_wv[l] = F(q++); sf_bv[l] = F(q++); sf_wo[l] = F(q++); sf_bo[l] = F(q++);
    sf_gf[l] = F(q++); sf_bf[l] = F(q++); sf_w1[l] = F(q++); sf_b1[l] = F(q++);
    sf_w2[l] = F(q++); sf_b2[l] = F(q++); sf_g[l]  = F(q++); sf_bs[l] = F(q++);
  }
  const float *og = F(q++), *obn = F(q++), *ow = F(q++), *ob = F(q++);

  // -------------------- workspace layout --------------------
  const size_t fB = (size_t)Brows;
  char* ws = (char*)d_ws;
  float*  h    = (float*)ws;                 // B x 512 f32 residual stream
  float*  pe   = h    + fB * 512;            // B x 512 f32
  float*  te   = pe   + fB * 512;            // B x 512 f32
  float*  r1   = te   + fB * 512;            // B x 512 f32 scratch
  bf16_t* ab1  = (bf16_t*)(r1 + fB * 512);   // B x 512 bf16 (A operand)
  bf16_t* ab2  = ab1  + fB * 512;            // B x 512 bf16
  bf16_t* gbuf = ab2  + fB * 512;            // B x 2048 bf16 (gelu out)
  bf16_t* pcb  = gbuf + fB * 2048;           // B x 1280 bf16
  bf16_t* xb   = pcb  + fB * 1280;           // B x 512 bf16
  bf16_t* e1b  = xb   + fB * 512;            // B x 512 bf16
  char*   wptr = (char*)(e1b + fB * 512);    // bf16 transposed weights

  auto prep = [&](const float* W, int K, int N) -> bf16_t* {
    bf16_t* dst = (bf16_t*)wptr;
    wptr += (size_t)K * N * sizeof(bf16_t);
    const int total = K * N;
    wtrans_kernel<<<(total + 255) / 256, 256, 0, stream>>>(W, dst, K, N);
    return dst;
  };

  bf16_t* wt_tw2 = prep(tw2, 512, 512);
  bf16_t* wt_pw1 = prep(pw1, 1280, 512);
  bf16_t* wt_pw2 = prep(pw2, 512, 512);
  bf16_t* wt_cw  = prep(cw, 512, 512);
  bf16_t *wt_crwv[3], *wt_crwo[3], *wt_crw1[3], *wt_crw2[3];
  bf16_t *wt_sfwv[3], *wt_sfwo[3], *wt_sfw1[3], *wt_sfw2[3];
  for (int l = 0; l < 3; ++l) {
    wt_crwv[l] = prep(cr_wv[l], 512, 512);
    wt_crwo[l] = prep(cr_wo[l], 512, 512);
    wt_crw1[l] = prep(cr_w1[l], 512, 2048);
    wt_crw2[l] = prep(cr_w2[l], 2048, 512);
    wt_sfwv[l] = prep(sf_wv[l], 512, 512);
    wt_sfwo[l] = prep(sf_wo[l], 512, 512);
    wt_sfw1[l] = prep(sf_w1[l], 512, 2048);
    wt_sfw2[l] = prep(sf_w2[l], 2048, 512);
  }
  bf16_t* wt_ow = prep(ow, 512, 512);

  auto gemm = [&](const bf16_t* Aop, const bf16_t* Wt, const float* bias,
                  const float* R, void* C, int Ntot, int K, int mode) {
    dim3 grid(Ntot / 128, Brows / 128);
    switch (mode) {
      case MODE_F32:       gemm_bf16<MODE_F32      ><<<grid, 256, 0, stream>>>(Aop, Wt, bias, R, C, Ntot, K); break;
      case MODE_BF16:      gemm_bf16<MODE_BF16     ><<<grid, 256, 0, stream>>>(Aop, Wt, bias, R, C, Ntot, K); break;
      case MODE_F32_RESID: gemm_bf16<MODE_F32_RESID><<<grid, 256, 0, stream>>>(Aop, Wt, bias, R, C, Ntot, K); break;
      default:             gemm_bf16<MODE_BF16_GELU><<<grid, 256, 0, stream>>>(Aop, Wt, bias, R, C, Ntot, K); break;
    }
  };
  auto ln = [&](const float* X, const float* gg, const float* bb, bf16_t* Y, int act) {
    ln_act_kernel<<<Brows / 8, 256, 0, stream>>>(X, gg, bb, Y, act);
  };

  // -------------------- forward pass --------------------
  // time embedding
  time_e1_kernel<<<(Brows * 512) / 256, 256, 0, stream>>>(t, tw1, tb1, e1b);
  gemm(e1b, wt_tw2, tb2, nullptr, te, 512, 512, MODE_F32);
  // protein embedding
  cvt_bf16_kernel<<<(Brows * 1280 + 255) / 256, 256, 0, stream>>>(pc, pcb, Brows * 1280);
  gemm(pcb, wt_pw1, pb1, nullptr, r1, 512, 1280, MODE_F32);
  ln(r1, pg, pbn, ab1, 1);
  gemm(ab1, wt_pw2, pb2, nullptr, pe, 512, 512, MODE_F32);
  // h = x @ comp.w + comp.b + te
  cvt_bf16_kernel<<<(Brows * 512 + 255) / 256, 256, 0, stream>>>(x, xb, Brows * 512);
  gemm(xb, wt_cw, cb, te, h, 512, 512, MODE_F32_RESID);

  for (int l = 0; l < 3; ++l) {
    // cross block (degenerate attention: value path only, sourced from pe)
    ln(pe, cr_gc[l], cr_bc[l], ab1, 0);
    gemm(ab1, wt_crwv[l], cr_bv[l], nullptr, ab2, 512, 512, MODE_BF16);
    gemm(ab2, wt_crwo[l], cr_bo[l], h, h, 512, 512, MODE_F32_RESID);
    ln(h, cr_gf[l], cr_bf[l], ab1, 0);
    gemm(ab1, wt_crw1[l], cr_b1[l], nullptr, gbuf, 2048, 512, MODE_BF16_GELU);
    gemm(gbuf, wt_crw2[l], cr_b2[l], h, h, 512, 2048, MODE_F32_RESID);
    // self block (value path sourced from h)
    ln(h, sf_g[l], sf_bs[l], ab1, 0);
    gemm(ab1, wt_sfwv[l], sf_bv[l], nullptr, ab2, 512, 512, MODE_BF16);
    gemm(ab2, wt_sfwo[l], sf_bo[l], h, h, 512, 512, MODE_F32_RESID);
    ln(h, sf_gf[l], sf_bf[l], ab1, 0);
    gemm(ab1, wt_sfw1[l], sf_b1[l], nullptr, gbuf, 2048, 512, MODE_BF16_GELU);
    gemm(gbuf, wt_sfw2[l], sf_b2[l], h, h, 512, 2048, MODE_F32_RESID);
  }

  // output head
  ln(h, og, obn, ab1, 1);
  gemm(ab1, wt_ow, ob, nullptr, (float*)d_out, 512, 512, MODE_F32);
}
